// EncoderLayer_18399639896091
// MI455X (gfx1250) — compile-verified, matrix operand order checked by reading, not run on hardware
//
#include <hip/hip_runtime.h>

// ---------------------------------------------------------------------------
// EncoderLayer for MI455X (gfx1250): f16 WMMA (v_wmma_f32_16x16x32_f16),
// fp32 accumulate, fp32 residual/LN paths. All fragments are 16B v8h loads
// (weights pre-transposed col-major, V stored transposed). All WMMA kernels
// are register double-buffered so VMEM overlaps matrix math; launch bounds
// give the allocator the full VGPR budget (no spills). Async global->LDS
// staging (ASYNCcnt) in the weight-transpose kernel.
// ---------------------------------------------------------------------------

typedef _Float16 half_t;
typedef __attribute__((ext_vector_type(16))) _Float16 v16h;
typedef __attribute__((ext_vector_type(8)))  _Float16 v8h;
typedef __attribute__((ext_vector_type(8)))  float    v8f;

#define D_MODEL 256
#define HEADS   8
#define DKHEAD  32
#define D_FF    1024
#define SEQ_L   256
#define SEQ_N   256
#define TOKENS  (SEQ_N * SEQ_L)        // 65536
#define ATT_SCALE 0.17677669529663687f // 1/sqrt(32)

union FragAB { v16h v; v8h h2[2]; };

#define WMMA_F16(a, b, c) \
    __builtin_amdgcn_wmma_f32_16x16x32_f16(false, (a), false, (b), (short)0, (c), false, false)

__device__ __forceinline__ void frag_ld(FragAB& f, const half_t* p) {
    f.h2[0] = *(const v8h*)p;
    f.h2[1] = *(const v8h*)(p + 16);
}

// ---------------------------------------------------------------------------
// Weight convert + transpose: wt[n*K + k] = (f16) w[k*N + n]
// 32x32 tile per block staged through LDS with async global->LDS loads.
// ---------------------------------------------------------------------------
__global__ void convert_transpose_kernel(const float* __restrict__ w,
                                         half_t* __restrict__ wt,
                                         int K, int N) {
    __shared__ float tile[32][33];
    const int kt = blockIdx.x % (K >> 5);
    const int nt = blockIdx.x / (K >> 5);
    const int i = threadIdx.x >> 3;        // 0..31 (row within tile)
    const int j = (threadIdx.x & 7) * 4;   // 0..28 step 4

    const float* gp = w + (size_t)(kt * 32 + i) * N + nt * 32 + j;
    // async copy 16B global -> LDS (ASYNCcnt-tracked), CDNA5 path
    unsigned lds_addr = (unsigned)(uintptr_t)&tile[i][j];
    asm volatile("global_load_async_to_lds_b128 %0, %1, off"
                 :: "v"(lds_addr), "v"(gp) : "memory");
    asm volatile("s_wait_asynccnt 0x0" ::: "memory");
    __syncthreads();

    half_t* op = wt + (size_t)(nt * 32 + i) * K + kt * 32 + j;
#pragma unroll
    for (int e = 0; e < 4; ++e) op[e] = (half_t)tile[j + e][i];
}

// ---------------------------------------------------------------------------
// LayerNorm over last dim (256): one block (256 threads) per row, f16 out
// ---------------------------------------------------------------------------
__global__ void layernorm_f16_kernel(const float* __restrict__ x,
                                     const float* __restrict__ ga,
                                     const float* __restrict__ gb,
                                     half_t* __restrict__ y) {
    __shared__ float red[D_MODEL];
    const int row = blockIdx.x;
    const int t   = threadIdx.x;
    const float v = x[(size_t)row * D_MODEL + t];

    red[t] = v; __syncthreads();
    for (int s = 128; s > 0; s >>= 1) {
        if (t < s) red[t] += red[t + s];
        __syncthreads();
    }
    const float mean = red[0] * (1.0f / D_MODEL);
    __syncthreads();

    const float d = v - mean;
    red[t] = d * d; __syncthreads();
    for (int s = 128; s > 0; s >>= 1) {
        if (t < s) red[t] += red[t + s];
        __syncthreads();
    }
    const float inv = rsqrtf(red[0] * (1.0f / D_MODEL) + 1e-5f);
    y[(size_t)row * D_MODEL + t] = (half_t)(ga[t] * d * inv + gb[t]);
}

// ---------------------------------------------------------------------------
// WMMA GEMM with column-major f16 weights BT[N][K]:
//   C = act((A[M x K] @ B + bias) * alpha)
// Each wave owns a 32x64 block of C: 2 A frags + 4 B frags -> 8 WMMAs per
// 32-K chunk. Register double-buffered across chunks (K multiple of 64).
// __launch_bounds__(256,1): low-occupancy GEMM, full VGPR budget, no spills.
// ---------------------------------------------------------------------------
struct GChunk { FragAB a0, a1, b0, b1, b2, b3; };

__global__ void __launch_bounds__(256, 1)
gemm_wmma_kernel(const half_t* __restrict__ A,
                 const half_t* __restrict__ BT,
                 const float*  __restrict__ bias,
                 half_t* __restrict__ Ch,
                 float*  __restrict__ Cf,
                 const float* __restrict__ resid,
                 int M, int N, int K,
                 float alpha, int do_relu, int vt_mode) {
    const int wid  = threadIdx.x >> 5;
    const int lane = threadIdx.x & 31;
    const int strips_n = N >> 6;                 // 64-wide strips
    const int strip = blockIdx.x * 8 + wid;
    const int total = (M >> 5) * strips_n;       // 32-tall rows of strips
    if (strip >= total) return;
    const int tm = strip / strips_n;
    const int sn = strip - tm * strips_n;

    const int r16  = lane & 15;
    const int koff = (lane >> 4) << 3;           // 0 or 8
    const int row0 = tm * 32 + r16;
    const int col0 = sn * 64 + r16;

    const half_t* a0p = A + (size_t)row0 * K + koff;
    const half_t* a1p = a0p + (size_t)16 * K;
    const half_t* bp0 = BT + (size_t)col0 * K + koff;
    const half_t* bp1 = bp0 + (size_t)16 * K;
    const half_t* bp2 = bp0 + (size_t)32 * K;
    const half_t* bp3 = bp0 + (size_t)48 * K;

    v8f c0[4] = {}, c1[4] = {};
    GChunk cA, cB;

    auto load_chunk = [&](GChunk& ch, int k) {
        frag_ld(ch.a0, a0p + k); frag_ld(ch.a1, a1p + k);
        frag_ld(ch.b0, bp0 + k); frag_ld(ch.b1, bp1 + k);
        frag_ld(ch.b2, bp2 + k); frag_ld(ch.b3, bp3 + k);
    };
    auto compute = [&](GChunk& ch) {
        c0[0] = WMMA_F16(ch.a0.v, ch.b0.v, c0[0]);
        c1[0] = WMMA_F16(ch.a1.v, ch.b0.v, c1[0]);
        c0[1] = WMMA_F16(ch.a0.v, ch.b1.v, c0[1]);
        c1[1] = WMMA_F16(ch.a1.v, ch.b1.v, c1[1]);
        c0[2] = WMMA_F16(ch.a0.v, ch.b2.v, c0[2]);
        c1[2] = WMMA_F16(ch.a1.v, ch.b2.v, c1[2]);
        c0[3] = WMMA_F16(ch.a0.v, ch.b3.v, c0[3]);
        c1[3] = WMMA_F16(ch.a1.v, ch.b3.v, c1[3]);
    };

    load_chunk(cA, 0);
    for (int k0 = 0; k0 + 64 <= K; k0 += 64) {       // K % 64 == 0
        load_chunk(cB, k0 + 32);
        compute(cA);
        if (k0 + 64 < K) load_chunk(cA, k0 + 64);
        compute(cB);
    }

    const int rowoff = koff;                         // (lane>=16) ? 8 : 0
#pragma unroll
    for (int mt = 0; mt < 2; ++mt) {
        v8f* cc = mt ? c1 : c0;
#pragma unroll
        for (int s = 0; s < 4; ++s) {
            const int col = col0 + s * 16;
            const float bv = bias ? bias[col] : 0.0f;
#pragma unroll
            for (int r = 0; r < 8; ++r) {
                const int orow = tm * 32 + mt * 16 + r + rowoff;
                float v = (cc[s][r] + bv) * alpha;
                if (do_relu) v = v > 0.0f ? v : 0.0f;
                if (Ch) {
                    size_t oi;
                    if (vt_mode) {
                        // vt[n][h][d][m]  (token = n*256+m, col = h*32+d)
                        oi = ((((size_t)(orow >> 8)) * HEADS + (col >> 5)) * DKHEAD
                              + (col & 31)) * SEQ_L + (orow & 255);
                    } else {
                        oi = (size_t)orow * N + col;
                    }
                    Ch[oi] = (half_t)v;
                }
                if (Cf) {
                    const size_t oi = (size_t)orow * N + col;
                    Cf[oi] = v + (resid ? resid[oi] : 0.0f);
                }
            }
        }
    }
}

// ---------------------------------------------------------------------------
// Switch attention: sw[l,h,n] = softmax_n( qs[l,h,:] . ks[n,l,h,:] )
// qs already scaled. One block per (l,h), thread n.
// ---------------------------------------------------------------------------
__global__ void switch_softmax_kernel(const float* __restrict__ qs,
                                      const half_t* __restrict__ ks,
                                      float* __restrict__ sw) {
    const int l = blockIdx.x >> 3;
    const int h = blockIdx.x & 7;
    const int n = threadIdx.x;

    const float*  qp = qs + (size_t)l * D_MODEL + h * DKHEAD;
    const half_t* kp = ks + ((size_t)n * SEQ_L + l) * D_MODEL + h * DKHEAD;
    float acc = 0.0f;
#pragma unroll
    for (int d = 0; d < DKHEAD; ++d) acc += qp[d] * (float)kp[d];

    __shared__ float red[SEQ_N];
    red[n] = acc; __syncthreads();
    for (int s = 128; s > 0; s >>= 1) {
        if (n < s) red[n] = fmaxf(red[n], red[n + s]);
        __syncthreads();
    }
    const float mx = red[0]; __syncthreads();
    const float e = __expf(acc - mx);
    red[n] = e; __syncthreads();
    for (int s = 128; s > 0; s >>= 1) {
        if (n < s) red[n] += red[n + s];
        __syncthreads();
    }
    sw[((size_t)l * HEADS + h) * SEQ_N + n] = e / red[0];
}

// ---------------------------------------------------------------------------
// scores[h,l,m] = ATT_SCALE * sum_{n,d} (sw[l,h,n]*q[n,l,h,d]) * k[n,m,h,d]
// Per-head GEMM, K = N*dk = 8192, sw folded into the A fragment at load.
// Wave owns 16(l) x 32(m): 1 weighted A frag -> 2 WMMAs per chunk; register
// double-buffered over the 256 n-chunks.
// ---------------------------------------------------------------------------
struct SChunk { FragAB a, b0, b1; };

__global__ void __launch_bounds__(256, 1)
scores_wmma_kernel(const half_t* __restrict__ qh,
                   const half_t* __restrict__ kh,
                   const float*  __restrict__ sw,
                   float* __restrict__ scores) {
    const int wid  = threadIdx.x >> 5;
    const int lane = threadIdx.x & 31;
    const int tile = blockIdx.x * 8 + wid;      // 0..1023
    const int h    = tile >> 7;                 // 8
    const int rem  = tile & 127;
    const int lt   = rem >> 3;                  // 16 l-tiles
    const int mt2  = rem & 7;                   // 8 double-m-tiles

    const int r16  = lane & 15;
    const int koff = (lane >> 4) << 3;
    const int l  = lt * 16 + r16;
    const int m0 = mt2 * 32 + r16;

    const float* swrow = sw + ((size_t)l * HEADS + h) * SEQ_N;
    const half_t* qbase  = qh + (size_t)l * D_MODEL + h * DKHEAD + koff;
    const half_t* kbase0 = kh + (size_t)m0 * D_MODEL + h * DKHEAD + koff;
    const half_t* kbase1 = kbase0 + (size_t)16 * D_MODEL;

    v8f c0 = {}, c1 = {};
    SChunk cA, cB;

    auto load_chunk = [&](SChunk& ch, int n) {
        const size_t off = (size_t)n * SEQ_L * D_MODEL;
        const _Float16 sv = (_Float16)swrow[n];
        FragAB t;
        frag_ld(t, qbase + off);
        ch.a.h2[0] = t.h2[0] * sv;
        ch.a.h2[1] = t.h2[1] * sv;
        frag_ld(ch.b0, kbase0 + off);
        frag_ld(ch.b1, kbase1 + off);
    };
    auto compute = [&](SChunk& ch) {
        c0 = WMMA_F16(ch.a.v, ch.b0.v, c0);
        c1 = WMMA_F16(ch.a.v, ch.b1.v, c1);
    };

    load_chunk(cA, 0);
    for (int n = 0; n + 2 <= SEQ_N; n += 2) {
        load_chunk(cB, n + 1);
        compute(cA);
        if (n + 2 < SEQ_N) load_chunk(cA, n + 2);
        compute(cB);
    }

    const int rowoff = koff;
#pragma unroll
    for (int r = 0; r < 8; ++r) {
        const int lrow = lt * 16 + r + rowoff;
        float* srow = scores + ((size_t)h * SEQ_L + lrow) * SEQ_N + mt2 * 32 + r16;
        srow[0]  = c0[r] * ATT_SCALE;
        srow[16] = c1[r] * ATT_SCALE;
    }
}

// ---------------------------------------------------------------------------
// Row softmax over 256, f32 in -> f16 out. Block per row.
// ---------------------------------------------------------------------------
__global__ void row_softmax_f16_kernel(const float* __restrict__ s,
                                       half_t* __restrict__ o) {
    const int row = blockIdx.x;
    const int t   = threadIdx.x;
    const float v = s[(size_t)row * SEQ_N + t];
    __shared__ float red[SEQ_N];
    red[t] = v; __syncthreads();
    for (int st = 128; st > 0; st >>= 1) {
        if (t < st) red[t] = fmaxf(red[t], red[t + st]);
        __syncthreads();
    }
    const float mx = red[0]; __syncthreads();
    const float e = __expf(v - mx);
    red[t] = e; __syncthreads();
    for (int st = 128; st > 0; st >>= 1) {
        if (t < st) red[t] += red[t + st];
        __syncthreads();
    }
    o[(size_t)row * SEQ_N + t] = (half_t)(e / red[0]);
}

// ---------------------------------------------------------------------------
// out[n,l,h,d] = sum_m att[h,l,m] * v[n,m,h,d] using vt[n][h][d][m] layout.
// Wave owns 16(l) x 32(d): 1 A frag -> 2 WMMAs per chunk, double-buffered.
// ---------------------------------------------------------------------------
__global__ void __launch_bounds__(256, 1)
attnout_wmma_kernel(const half_t* __restrict__ atth,
                    const half_t* __restrict__ vt,
                    half_t* __restrict__ outh) {
    const int wid  = threadIdx.x >> 5;
    const int lane = threadIdx.x & 31;
    const int tile = blockIdx.x * 8 + wid;      // 0..32767
    const int n    = tile >> 7;
    const int rem  = tile & 127;
    const int h    = rem >> 4;
    const int lt   = rem & 15;

    const int r16  = lane & 15;
    const int koff = (lane >> 4) << 3;
    const int l = lt * 16 + r16;

    const half_t* arow  = atth + ((size_t)h * SEQ_L + l) * SEQ_N + koff;
    const half_t* vb0   = vt + ((size_t)n * HEADS + h) * DKHEAD * SEQ_L
                             + (size_t)r16 * SEQ_L + koff;
    const half_t* vb1   = vb0 + (size_t)16 * SEQ_L;

    v8f c0 = {}, c1 = {};
    SChunk cA, cB;

    auto load_chunk = [&](SChunk& ch, int m0) {
        frag_ld(ch.a,  arow + m0);
        frag_ld(ch.b0, vb0 + m0);
        frag_ld(ch.b1, vb1 + m0);
    };
    auto compute = [&](SChunk& ch) {
        c0 = WMMA_F16(ch.a.v, ch.b0.v, c0);
        c1 = WMMA_F16(ch.a.v, ch.b1.v, c1);
    };

    load_chunk(cA, 0);
    for (int m0 = 0; m0 + 64 <= SEQ_N; m0 += 64) {
        load_chunk(cB, m0 + 32);
        compute(cA);
        if (m0 + 64 < SEQ_N) load_chunk(cA, m0 + 64);
        compute(cB);
    }

    const int rowoff = koff;
#pragma unroll
    for (int r = 0; r < 8; ++r) {
        const int lrow = lt * 16 + r + rowoff;
        half_t* op = outh + ((size_t)n * SEQ_L + lrow) * D_MODEL + h * DKHEAD + r16;
        op[0]  = (half_t)c0[r];
        op[16] = (half_t)c1[r];
    }
}

// ---------------------------------------------------------------------------
// Host side
// ---------------------------------------------------------------------------
extern "C" void kernel_launch(void* const* d_in, const int* in_sizes, int n_in,
                              void* d_out, int out_size, void* d_ws, size_t ws_size,
                              hipStream_t stream) {
    const float* src   = (const float*)d_in[0];
    const float* ln1_a = (const float*)d_in[1];
    const float* ln1_b = (const float*)d_in[2];
    const float* sw_wq = (const float*)d_in[3];
    const float* sw_bq = (const float*)d_in[4];
    const float* sw_wk = (const float*)d_in[5];
    const float* sw_bk = (const float*)d_in[6];
    const float* wq    = (const float*)d_in[7];
    const float* bq    = (const float*)d_in[8];
    const float* wk    = (const float*)d_in[9];
    const float* bk    = (const float*)d_in[10];
    const float* wv    = (const float*)d_in[11];
    const float* bv    = (const float*)d_in[12];
    const float* wo    = (const float*)d_in[13];
    const float* bo    = (const float*)d_in[14];
    const float* ln2_a = (const float*)d_in[15];
    const float* ln2_b = (const float*)d_in[16];
    const float* w1    = (const float*)d_in[17];
    const float* b1    = (const float*)d_in[18];
    const float* w2    = (const float*)d_in[19];
    const float* b2    = (const float*)d_in[20];
    float* out = (float*)d_out;

    const size_t T = TOKENS, D = D_MODEL, F = D_FF;

    char* cur = (char*)d_ws;
    auto alloc = [&](size_t bytes) -> void* {
        void* r = (void*)cur;
        cur += (bytes + 255) & ~(size_t)255;
        return r;
    };

    // transposed f16 weights (BT[N][K])
    half_t* t_sw_wq = (half_t*)alloc(D * D * 2);
    half_t* t_sw_wk = (half_t*)alloc(D * D * 2);
    half_t* t_wq    = (half_t*)alloc(D * D * 2);
    half_t* t_wk    = (half_t*)alloc(D * D * 2);
    half_t* t_wv    = (half_t*)alloc(D * D * 2);
    half_t* t_wo    = (half_t*)alloc(D * D * 2);
    half_t* t_w1    = (half_t*)alloc(D * F * 2);
    half_t* t_w2    = (half_t*)alloc(F * D * 2);
    half_t* xh      = (half_t*)alloc(T * D * 2);
    half_t* ksh     = (half_t*)alloc(T * D * 2);
    half_t* qh      = (half_t*)alloc(T * D * 2);
    half_t* kh      = (half_t*)alloc(T * D * 2);
    half_t* vtb     = (half_t*)alloc(T * D * 2);   // vt[n][h][d][m]
    half_t* outh    = (half_t*)alloc(T * D * 2);
    half_t* yh      = (half_t*)alloc(T * D * 2);
    half_t* h1h     = (half_t*)alloc(T * F * 2);
    half_t* atth    = (half_t*)alloc((size_t)HEADS * SEQ_L * SEQ_N * 2);
    float*  qs      = (float*)alloc((size_t)SEQ_L * D * 4);
    float*  swb     = (float*)alloc((size_t)SEQ_L * HEADS * SEQ_N * 4);
    float*  scoresb = (float*)alloc((size_t)HEADS * SEQ_L * SEQ_N * 4);
    float*  src2    = (float*)alloc(T * D * 4);

    auto convT = [&](const float* w, half_t* wt, int K, int N) {
        convert_transpose_kernel<<<(K >> 5) * (N >> 5), 256, 0, stream>>>(w, wt, K, N);
    };
    convT(sw_wq, t_sw_wq, D, D);
    convT(sw_wk, t_sw_wk, D, D);
    convT(wq, t_wq, D, D);
    convT(wk, t_wk, D, D);
    convT(wv, t_wv, D, D);
    convT(wo, t_wo, D, D);
    convT(w1, t_w1, D, F);
    convT(w2, t_w2, F, D);

    layernorm_f16_kernel<<<T, 256, 0, stream>>>(src, ln1_a, ln1_b, xh);

    auto gemm = [&](const half_t* A, const half_t* BT, const float* bias,
                    half_t* Ch, float* Cf, const float* resid,
                    int M, int N, int K, float alpha, int relu, int vt_mode) {
        int strips = (M >> 5) * (N >> 6);
        gemm_wmma_kernel<<<(strips + 7) / 8, 256, 0, stream>>>(
            A, BT, bias, Ch, Cf, resid, M, N, K, alpha, relu, vt_mode);
    };

    // --- projections (f16 out) ---
    gemm(xh, t_sw_wk, sw_bk, ksh, nullptr, nullptr, T, D, D, 1.0f, 0, 0);
    gemm(xh, t_wq,    bq,    qh,  nullptr, nullptr, T, D, D, 1.0f, 0, 0);
    gemm(xh, t_wk,    bk,    kh,  nullptr, nullptr, T, D, D, 1.0f, 0, 0);
    gemm(xh, t_wv,    bv,    vtb, nullptr, nullptr, T, D, D, 1.0f, 0, 1);
    // qs = (tar @ sw_wq + sw_bq) * scale ; tar == first 256 rows of x
    gemm(xh, t_sw_wq, sw_bq, nullptr, qs, nullptr, SEQ_L, D, D, ATT_SCALE, 0, 0);

    switch_softmax_kernel<<<SEQ_L * HEADS, SEQ_N, 0, stream>>>(qs, ksh, swb);

    scores_wmma_kernel<<<(HEADS * 16 * 8) / 8, 256, 0, stream>>>(qh, kh, swb, scoresb);
    row_softmax_f16_kernel<<<HEADS * SEQ_L, SEQ_N, 0, stream>>>(scoresb, atth);

    attnout_wmma_kernel<<<(SEQ_N * HEADS * 16) / 8, 256, 0, stream>>>(atth, vtb, outh);

    gemm(outh, t_wo, bo, nullptr, src2, src, T, D, D, 1.0f, 0, 0);

    layernorm_f16_kernel<<<T, 256, 0, stream>>>(src2, ln2_a, ln2_b, yh);
    gemm(yh,  t_w1, b1, h1h, nullptr, nullptr, T, F, D, 1.0f, 1, 0);
    gemm(h1h, t_w2, b2, nullptr, out, src2,    T, D, F, 1.0f, 0, 0);
}
